// LVCBlock_24532853195231
// MI455X (gfx1250) — matrix operand order repaired and checked by reading, not compiled, not run on hardware
//
#include <hip/hip_runtime.h>
#include <hip/hip_bf16.h>

#define T_INQ  4096
#define T_OUT  32768
#define NB     2
#define NC     64
#define NO     128
#define NKL    128
#define NCOND  80
#define NH     64
#define LWCH   98304   // C*O*K*L predicted-kernel channels

typedef __attribute__((ext_vector_type(16))) __bf16 v16bf;
typedef __attribute__((ext_vector_type(8)))  float  v8f;

__device__ __forceinline__ v8f wmma_bf16(v16bf a, v16bf b, v8f c) {
  return __builtin_amdgcn_wmma_f32_16x16x32_bf16(false, a, false, b, (short)0, c,
                                                 false, false);
}

// 16-bit A-matrix 16x32 lane/VGPR layout (ISA 7.12.2).
__device__ __forceinline__ int frag_a_k(int lane, int e) {
  int half = (lane >> 4) & 1;
  int j = e >> 1, p = e & 1;
  int base = (j < 4) ? (2 * j) : (16 + 2 * (j - 4));
  return base + 8 * half + p;
}
// 16-bit B-matrix 32x16 layout.
__device__ __forceinline__ int frag_b_k(int lane, int e) {
  int half = (lane >> 4) & 1;
  int j = e >> 1, p = e & 1;
  int base = (j < 4) ? (2 * j) : (16 + 2 * (j - 4));
  return base + 8 * half + p;
}
__device__ __forceinline__ float lrelu(float x, float a) { return x >= 0.f ? x : a * x; }

// ---------------- kernel predictor trunk (VALU, tiny) ----------------

__global__ __launch_bounds__(256) void kpin_kernel(
    const float* __restrict__ c, const float* __restrict__ w,
    const float* __restrict__ bb, float* __restrict__ h0) {
  int i = blockIdx.x * 256 + threadIdx.x;              // NB*128*64
  int ch = i & 63, t = (i >> 6) & 127, b = i >> 13;
  float acc = bb[ch];
  for (int k = 0; k < 5; ++k) {
    int tt = t + k - 2;
    if (tt < 0 || tt >= NKL) continue;
    const float* cp = c + ((size_t)b * NKL + tt) * NCOND;
    const float* wp = w + (size_t)k * NCOND * NH + ch;
    for (int j = 0; j < NCOND; ++j) acc += cp[j] * wp[j * NH];
  }
  h0[i] = lrelu(acc, 0.1f);
}

__global__ __launch_bounds__(256) void kpres_kernel(
    const float* __restrict__ in, const float* __restrict__ w,
    const float* __restrict__ bb, float* __restrict__ out) {
  int i = blockIdx.x * 256 + threadIdx.x;              // NB*128*64
  int ch = i & 63, t = (i >> 6) & 127, b = i >> 13;
  float acc = bb[ch];
  for (int k = 0; k < 3; ++k) {
    int tt = t + k - 1;
    if (tt < 0 || tt >= NKL) continue;
    const float* ip = in + ((size_t)b * NKL + tt) * NH;
    const float* wp = w + (size_t)k * NH * NH + ch;
    for (int j = 0; j < NH; ++j) acc += ip[j] * wp[j * NH];
  }
  out[i] = lrelu(acc, 0.1f);
}

__global__ __launch_bounds__(256) void kpsum_kernel(
    const float* __restrict__ h0, const float* __restrict__ r,
    float* __restrict__ hf, unsigned short* __restrict__ hfb_u) {
  int i = blockIdx.x * 256 + threadIdx.x;
  float v = h0[i] + r[i];
  hf[i] = v;
  reinterpret_cast<__bf16*>(hfb_u)[i] = (__bf16)v;
}

// bias head: (B,128,512) -> bias[b][l][o][w]
__global__ __launch_bounds__(256) void kpbias_kernel(
    const float* __restrict__ hf, const float* __restrict__ w,
    const float* __restrict__ bb, float* __restrict__ biasb) {
  int i = blockIdx.x * 256 + threadIdx.x;              // NB*128*512
  int ch = i & 511, t = (i >> 9) & 127, b = i >> 16;
  float acc = bb[ch];
  for (int k = 0; k < 3; ++k) {
    int tt = t + k - 1;
    if (tt < 0 || tt >= NKL) continue;
    const float* hp = hf + ((size_t)b * NKL + tt) * NH;
    const float* wp = w + (size_t)k * NH * 512 + ch;
    for (int j = 0; j < NH; ++j) acc += hp[j] * wp[j * 512];
  }
  int l = ch >> 7, o = ch & 127;
  biasb[(((size_t)b * 4 + l) * NO + o) * NKL + t] = acc;
}

// ------------- k head: WMMA GEMM M=98304, N=256, K=192, scatter to ker -------------
// ker layout: [b][l][w][kk = tap*64+ci][o]  (bf16)
__global__ __launch_bounds__(256, 1) void kpk_wmma_kernel(
    const float* __restrict__ kw, const float* __restrict__ kb,
    const unsigned short* __restrict__ hfb_u, unsigned short* __restrict__ ker_u) {
  __shared__ __bf16 s_hf[NB * 130 * NH];               // zero-padded (ww in [-1,128])
  const __bf16* hf = reinterpret_cast<const __bf16*>(hfb_u);
  __bf16* ker = reinterpret_cast<__bf16*>(ker_u);

  for (int idx = threadIdx.x; idx < NB * 130 * NH; idx += 256) {
    int hh = idx & 63, row = idx >> 6;
    int b = row / 130, ww = row - b * 130 - 1;
    __bf16 v = (__bf16)0.f;
    if (ww >= 0 && ww < NKL) v = hf[((size_t)b * NKL + ww) * NH + hh];
    s_hf[idx] = v;
  }
  __syncthreads();

  const int lane = threadIdx.x & 31;
  const int wv = threadIdx.x >> 5;
  const int ch0 = blockIdx.x * 128 + wv * 16;
  const int ch = ch0 + (lane & 15);
  const int half = (lane >> 4) & 1;
  const int ncol = lane & 15;

  v16bf a[6];
#pragma unroll
  for (int ks = 0; ks < 6; ++ks)
#pragma unroll
    for (int e = 0; e < 16; ++e) {
      int kk = ks * 32 + frag_a_k(lane, e);
      a[ks][e] = (__bf16)kw[(size_t)kk * LWCH + ch];
    }

  // per-lane D scatter offsets + bias, hoisted out of the nt loop (32-bit)
  int sbase[8];
  float kb8[8];
#pragma unroll
  for (int v = 0; v < 8; ++v) {
    int chd = ch0 + v + 8 * half;
    kb8[v] = kb[chd];
    int ktap = chd % 3;
    int rest = chd / 3;
    int o = rest & 127;
    int rest2 = rest >> 7;
    int ci = rest2 & 63;
    int l = rest2 >> 6;
    sbase[v] = l * (NKL * 192 * NO) + (ktap * 64 + ci) * NO + o;
  }

  for (int nt = 0; nt < 16; ++nt) {
    int n = nt * 16 + ncol;
    int b = n >> 7, w = n & 127;
    int srow = (b * 130 + w) * 64;                     // s_hf index of (b, ww = w-1)
    v8f acc = {};
#pragma unroll
    for (int ks = 0; ks < 6; ++ks) {
      v16bf bf;
#pragma unroll
      for (int e = 0; e < 16; ++e) {
        int kk = ks * 32 + frag_b_k(lane, e);
        int tap = kk >> 6, hh = kk & 63;
        bf[e] = s_hf[srow + tap * 64 + hh];
      }
      acc = wmma_bf16(a[ks], bf, acc);
    }
    int wb = b * (4 * NKL * 192 * NO) + w * (192 * NO);
#pragma unroll
    for (int v = 0; v < 8; ++v)
      ker[sbase[v] + wb] = (__bf16)(acc[v] + kb8[v]);
  }
}

// ---------------- upsample transpose conv (stride 8, K=16, SAME pad_a=11) ----------------
__global__ __launch_bounds__(256) void upsample_kernel(
    const float* __restrict__ x, const float* __restrict__ uw,
    const float* __restrict__ ub, float* __restrict__ h) {
  size_t i = (size_t)blockIdx.x * 256 + threadIdx.x;   // NB*T_OUT*NC
  int co = (int)(i & 63);
  size_t bt = i >> 6;
  int t = (int)(bt & (T_OUT - 1));
  int b = (int)(bt >> 15);
  int k0 = ((11 - t) % 8 + 8) % 8;
  float acc = ub[co];
  for (int tap = k0; tap < 16; tap += 8) {
    int j = t - 11 + tap;
    if (j < 0) continue;
    int u = j >> 3;
    if (u >= T_INQ) continue;
    const float* xp = x + ((size_t)b * T_INQ + u) * NC;
    const float* wp = uw + (size_t)tap * NC * NC + co;
    for (int ci = 0; ci < NC; ++ci) acc += lrelu(xp[ci], 0.2f) * wp[ci * NC];
  }
  h[i] = acc;
}

// ---------------- per-layer dilated conv (WMMA), writes bf16 y1 ----------------
__global__ __launch_bounds__(256, 1) void conv_wmma_kernel(
    const float* __restrict__ h, const float* __restrict__ w,
    const float* __restrict__ bias, int dil, unsigned short* __restrict__ y1_u) {
  __shared__ __bf16 s_h[86 * NC];                      // 32 + 2*dil rows, dil<=27
  __bf16* y1 = reinterpret_cast<__bf16*>(y1_u);
  const int b = blockIdx.x >> 10;
  const int t0b = (blockIdx.x & 1023) * 32;
  const int span = 32 + 2 * dil;
  for (int idx = threadIdx.x; idx < span * NC; idx += 256) {
    int p = idx >> 6, ci = idx & 63;
    int tt = t0b - dil + p;
    float v = 0.f;
    if (tt >= 0 && tt < T_OUT) v = lrelu(h[((size_t)b * T_OUT + tt) * NC + ci], 0.2f);
    s_h[idx] = (__bf16)v;
  }
  __syncthreads();

  const int lane = threadIdx.x & 31;
  const int wv = threadIdx.x >> 5;
  const int mt = wv >> 1;
  const int half = (lane >> 4) & 1;
  const int ncol = lane & 15;
  const int co_a = mt * 16 + (lane & 15);
  const int tloc = (wv & 1) * 16 + ncol;               // time within block chunk

  v16bf a[6];
#pragma unroll
  for (int ks = 0; ks < 6; ++ks)
#pragma unroll
    for (int e = 0; e < 16; ++e) {
      int kk = ks * 32 + frag_a_k(lane, e);
      int k = kk >> 6, ci = kk & 63;
      a[ks][e] = (__bf16)w[((size_t)k * NC + ci) * NC + co_a];
    }

  v8f acc = {};
#pragma unroll
  for (int ks = 0; ks < 6; ++ks) {
    v16bf bf;
#pragma unroll
    for (int e = 0; e < 16; ++e) {
      int kk = ks * 32 + frag_b_k(lane, e);
      int k = kk >> 6, ci = kk & 63;
      bf[e] = s_h[(tloc + k * dil) * NC + ci];         // p = tloc + k*dil
    }
    acc = wmma_bf16(a[ks], bf, acc);
  }

  const int t = t0b + tloc;
  __bf16* orow = y1 + ((size_t)b * T_OUT + t) * NC + mt * 16 + 8 * half;
  const float* bp = bias + mt * 16 + 8 * half;
#pragma unroll
  for (int v = 0; v < 8; ++v)
    orow[v] = (__bf16)lrelu(acc[v] + bp[v], 0.2f);
}

// ---------------- LVC: per (b, window) GEMM 128x256x192 with window-specific A ----------------
__global__ __launch_bounds__(256, 1) void lvc_wmma_kernel(
    const unsigned short* __restrict__ y1_u, const unsigned short* __restrict__ ker_u,
    const float* __restrict__ biasb, int layer, float* __restrict__ y2) {
  __shared__ alignas(16) __bf16 s_y[258 * NC];
  const __bf16* y1 = reinterpret_cast<const __bf16*>(y1_u);
  const __bf16* ker = reinterpret_cast<const __bf16*>(ker_u);
  const int b = blockIdx.x >> 7;
  const int w = blockIdx.x & 127;
  const int tid = threadIdx.x;

  // Interior rows p=1..256 (always in-bounds): async DMA global -> LDS, 16B/lane.
  {
    unsigned lds_off = (unsigned)(size_t)(&s_y[NC]) + (unsigned)tid * 16u;
    unsigned long long gaddr =
        (unsigned long long)(size_t)(y1 + ((size_t)b * T_OUT + (size_t)w * 256) * NC) +
        (unsigned long long)tid * 16ull;
#pragma unroll
    for (int j = 0; j < 8; ++j) {
      asm volatile("global_load_async_to_lds_b128 %0, %1, off"
                   :: "v"(lds_off), "v"(gaddr)
                   : "memory");
      lds_off += 256u * 16u;
      gaddr += 256ull * 16ull;
    }
  }
  // Boundary rows p=0 and p=257 (may be out of range at sequence edges).
  if (tid < 2 * NC) {
    int p = (tid < NC) ? 0 : 257;
    int ci = tid & 63;
    int tg = w * 256 + p - 1;
    __bf16 v = (__bf16)0.f;
    if (tg >= 0 && tg < T_OUT) v = y1[((size_t)b * T_OUT + tg) * NC + ci];
    s_y[p * NC + ci] = v;
  }
  asm volatile("s_wait_asynccnt 0x0" ::: "memory");
  __syncthreads();

  const int lane = tid & 31;
  const int wv = tid >> 5;                             // 8 waves = 8 M-tiles of O=128
  const int half = (lane >> 4) & 1;
  const int ncol = lane & 15;
  const int o_a = wv * 16 + (lane & 15);
  const __bf16* kslice = ker + (((size_t)b * 4 + layer) * NKL + w) * 192 * NO;

  v16bf a[6];
#pragma unroll
  for (int ks = 0; ks < 6; ++ks)
#pragma unroll
    for (int e = 0; e < 16; ++e) {
      int kk = ks * 32 + frag_a_k(lane, e);
      a[ks][e] = kslice[kk * NO + o_a];
    }

  // bias for this lane's 8 contiguous output channels, hoisted
  const int obase = wv * 16 + 8 * half;
  float bias8[8];
#pragma unroll
  for (int v = 0; v < 8; ++v)
    bias8[v] = biasb[(((size_t)b * 4 + layer) * NO + obase + v) * NKL + w];

  for (int nt = 0; nt < 16; ++nt) {
    int tl = nt * 16 + ncol;
    v8f acc = {};
#pragma unroll
    for (int ks = 0; ks < 6; ++ks) {
      v16bf bf;
#pragma unroll
      for (int e = 0; e < 16; ++e) {
        int kk = ks * 32 + frag_b_k(lane, e);
        int k = kk >> 6, ci = kk & 63;
        bf[e] = s_y[(tl + k) * NC + ci];
      }
      acc = wmma_bf16(a[ks], bf, acc);
    }
    int t = w * 256 + tl;
    float4* p = reinterpret_cast<float4*>(y2 + ((size_t)b * T_OUT + t) * NO + obase);
    float4 r0, r1;
    r0.x = acc[0] + bias8[0]; r0.y = acc[1] + bias8[1];
    r0.z = acc[2] + bias8[2]; r0.w = acc[3] + bias8[3];
    r1.x = acc[4] + bias8[4]; r1.y = acc[5] + bias8[5];
    r1.z = acc[6] + bias8[6]; r1.w = acc[7] + bias8[7];
    p[0] = r0;
    p[1] = r1;
  }
}

// ---------------- gated residual ----------------
__global__ __launch_bounds__(256) void gate_kernel(
    const float* __restrict__ y2, float* __restrict__ h) {
  size_t i = (size_t)blockIdx.x * 256 + threadIdx.x;   // NB*T_OUT*NC
  int c = (int)(i & 63);
  size_t bt = i >> 6;
  float s = y2[bt * NO + c];
  float tv = y2[bt * NO + c + NC];
  float sig = 1.f / (1.f + __expf(-s));
  h[i] += sig * tanhf(tv);
}

extern "C" void kernel_launch(void* const* d_in, const int* in_sizes, int n_in,
                              void* d_out, int out_size, void* d_ws, size_t ws_size,
                              hipStream_t stream) {
  const float* x       = (const float*)d_in[0];
  const float* c       = (const float*)d_in[1];
  const float* up_w    = (const float*)d_in[2];
  const float* up_b    = (const float*)d_in[3];
  const float* conv_w  = (const float*)d_in[4];
  const float* conv_b  = (const float*)d_in[5];
  const float* kp_in_w = (const float*)d_in[6];
  const float* kp_in_b = (const float*)d_in[7];
  const float* kp_res_w= (const float*)d_in[8];
  const float* kp_res_b= (const float*)d_in[9];
  const float* kp_k_w  = (const float*)d_in[10];
  const float* kp_k_b  = (const float*)d_in[11];
  const float* kp_b_w  = (const float*)d_in[12];
  const float* kp_b_b  = (const float*)d_in[13];
  (void)in_sizes; (void)n_in; (void)out_size; (void)ws_size;

  char* ws = (char*)d_ws;
  size_t ofs = 0;
  auto carve = [&](size_t bytes) -> void* {
    void* p = (void*)(ws + ofs);
    ofs += (bytes + 255) & ~(size_t)255;
    return p;
  };
  const size_t nHT = (size_t)NB * T_OUT * NC;          // 4,194,304
  float* h            = (float*)carve(nHT * 4);
  float* y2           = (float*)carve((size_t)NB * T_OUT * NO * 4);
  float* biasb        = (float*)carve((size_t)NB * 4 * NO * NKL * 4);
  float* h0           = (float*)carve((size_t)NB * NKL * NH * 4);
  float* ra           = (float*)carve((size_t)NB * NKL * NH * 4);
  float* rb           = (float*)carve((size_t)NB * NKL * NH * 4);
  float* hf           = (float*)carve((size_t)NB * NKL * NH * 4);
  unsigned short* y1  = (unsigned short*)carve(nHT * 2);
  unsigned short* hfb = (unsigned short*)carve((size_t)NB * NKL * NH * 2);
  unsigned short* ker = (unsigned short*)carve((size_t)NB * 4 * NKL * 192 * NO * 2);

  // kernel predictor trunk
  kpin_kernel<<<64, 256, 0, stream>>>(c, kp_in_w, kp_in_b, h0);
  float* rr[2] = {ra, rb};
  for (int j = 0; j < 6; ++j) {
    const float* in = (j == 0) ? h0 : rr[(j + 1) & 1];
    kpres_kernel<<<64, 256, 0, stream>>>(in, kp_res_w + (size_t)j * 3 * NH * NH,
                                         kp_res_b + (size_t)j * NH, rr[j & 1]);
  }
  kpsum_kernel<<<64, 256, 0, stream>>>(h0, rr[1], hf, hfb);
  kpbias_kernel<<<512, 256, 0, stream>>>(hf, kp_b_w, kp_b_b, biasb);
  kpk_wmma_kernel<<<LWCH / 128, 256, 0, stream>>>(kp_k_w, kp_k_b, hfb, ker);

  // main path
  upsample_kernel<<<(int)(nHT / 256), 256, 0, stream>>>(x, up_w, up_b, h);
  int dil = 1;
  for (int i = 0; i < 4; ++i) {
    conv_wmma_kernel<<<NB * (T_OUT / 32), 256, 0, stream>>>(
        h, conv_w + (size_t)i * 3 * NC * NC, conv_b + (size_t)i * NC, dil, y1);
    lvc_wmma_kernel<<<NB * NKL, 256, 0, stream>>>(y1, ker, biasb, i, y2);
    gate_kernel<<<(int)(nHT / 256), 256, 0, stream>>>(y2, h);
    dil *= 3;
  }

  hipMemcpyAsync(d_out, h, nHT * sizeof(float), hipMemcpyDeviceToDevice, stream);
}
